// GRUModelModify_58806692217520
// MI455X (gfx1250) — compile-verified, matrix operand order checked by reading, not run on hardware
//
#include <hip/hip_runtime.h>
#include <math.h>

// Problem constants (match reference)
#define B_ 128
#define S_ 2048
#define H_ 128
#define L_ 2
#define O_ 2

typedef __attribute__((ext_vector_type(16))) __bf16       v16bf;
typedef __attribute__((ext_vector_type(8)))  __bf16       v8bf;
typedef __attribute__((ext_vector_type(8)))  float        v8f;
typedef __attribute__((ext_vector_type(4)))  unsigned int v4u;
typedef __attribute__((ext_vector_type(8)))  int          i32x8;
typedef __attribute__((ext_vector_type(4)))  int          i32x4;

union Frag {
    v16bf v;
    v4u   u[2];
};

// fast tanh: 1 - 2/(exp(2x)+1); exp2 saturates correctly at +/-inf
__device__ __forceinline__ float tanh_fast(float x) {
    float e = __builtin_amdgcn_exp2f(x * 2.88539008177793f);   // exp(2x)
    return 1.0f - 2.0f * __builtin_amdgcn_rcpf(e + 1.0f);
}
__device__ __forceinline__ float sigmoid_fast(float x) {
    float e = __builtin_amdgcn_exp2f(x * -1.44269504088896f);  // exp(-x)
    return __builtin_amdgcn_rcpf(1.0f + e);
}

__global__ void zero_out_kernel(float* out) { out[0] = 0.0f; }

#if __has_builtin(__builtin_amdgcn_tensor_load_to_lds)
#define USE_TDM 1
#else
#define USE_TDM 0
#endif

#if USE_TDM
// Issue one TDM 2D tile load: 16 rows x 128 cols of f32, row stride S_*H_
// elements, from global gaddr into LDS byte offset ldsOff.
// This toolchain exposes the 6-arg builtin:
//   (uint32x4 g0, int32x8 g1, int32x4 g2, int32x4 g3, int32x8 g4, i32 cpol)
__device__ __forceinline__ void tdm_load_tile(unsigned long long gaddr,
                                              unsigned int ldsOff) {
    v4u g0;
    g0[0] = 1u;                                        // count=1, user D#
    g0[1] = ldsOff;                                    // lds_addr (bytes)
    g0[2] = (unsigned int)(gaddr & 0xFFFFFFFFu);       // global_addr[31:0]
    g0[3] = (unsigned int)((gaddr >> 32) & 0x01FFFFFFu)
          | (2u << 30);                                // addr[56:32] | type=2
    i32x8 g1;
    g1[0] = (int)(2u << 16);        // data_size=2 (4B); wg_mask=0
    g1[1] = (int)(128u << 16);      // tensor_dim0[15:0]=128 (bits 63:48)
    g1[2] = (int)(16u  << 16);      // tensor_dim1[15:0]=16  (bits 95:80)
    g1[3] = (int)(128u << 16);      // tile_dim0=128         (bits 127:112)
    g1[4] = 16;                     // tile_dim1=16, tile_dim2=0
    g1[5] = (int)(S_ * H_);         // tensor_dim0_stride[31:0] (elements)
    g1[6] = 0;                      // stride hi / dim1_stride lo
    g1[7] = 0;
    i32x4 gz4 = {0, 0, 0, 0};               // groups 2/3 unused (2D tensor)
    i32x8 gz8 = {0, 0, 0, 0, 0, 0, 0, 0};   // extra group (unused)
    __builtin_amdgcn_tensor_load_to_lds(g0, g1, gz4, gz4, gz8, 0);
}
#endif

// One workgroup per 16-row batch tile. 256 threads = 8 waves (wave32).
// Wave w computes gate N-tiles {w, w+8}: m1 cols [w*16,..) and m2 cols
// [128+w*16,..), so the gated state update is wave-local VALU.
__global__ __launch_bounds__(256) void sru_seq_kernel(
    const float* __restrict__ x_data,   // [B,S,H] f32
    const int*   __restrict__ x_length, // [B]
    const float* __restrict__ x_label,  // [B]
    const float* __restrict__ Wx2h,     // [L,2H,H] f32
    const float* __restrict__ Wfc,      // [L,O,H] f32
    float*       __restrict__ out)      // scalar
{
    __shared__ __align__(16) __bf16 wp[2 * 16 * 4 * 32 * 16]; // 128 KB weights
    __shared__ __align__(16) __bf16 xbuf[16 * 128];           // layer-0 A (bf16)
    __shared__ __align__(16) __bf16 hbuf[16 * 128];           // layer-1 A / fc src
    __shared__ __align__(16) float  wfc[O_ * H_];             // last-layer fc
#if USE_TDM
    __shared__ __align__(16) float  xstage[2][16 * 128];      // TDM double buffer
#endif

    const int tid  = threadIdx.x;
    const int lane = tid & 31;
    const int wv   = tid >> 5;           // wave id 0..7
    const int row0 = blockIdx.x * 16;    // batch tile base

    // ---------- one-time: prepack Wx2h (f32 -> bf16) into B-fragment order ----
    for (int idx = tid; idx < 65536; idx += 256) {
        const int e  = idx & 15;
        const int ln = (idx >> 4) & 31;
        const int kk = (idx >> 9) & 3;
        const int nt = (idx >> 11) & 15;
        const int l  = (idx >> 15) & 1;
        const int ng   = nt * 16 + (ln & 15);                    // gate column
        const int krel = (e < 8 ? e : e + 8) + ((ln >> 4) & 1) * 8;
        const int kg   = kk * 32 + krel;                         // K index
        wp[idx] = (__bf16)Wx2h[((size_t)(l * 256 + ng)) * H_ + kg];
    }
    wfc[tid] = Wfc[(L_ - 1) * O_ * H_ + tid];
    __syncthreads();

    // ---------- persistent per-thread state ----------------------------------
    float s1[L_][8], s2[L_][8];
#pragma unroll
    for (int l = 0; l < L_; ++l)
#pragma unroll
        for (int i = 0; i < 8; ++i) { s1[l][i] = 0.0f; s2[l][i] = 0.0f; }

    float lossAcc = 0.0f;
    int   len = 0; float label = 0.0f;
    if (tid < 16) { len = x_length[row0 + tid]; label = x_label[row0 + tid]; }

    // fragment addressing (ISA 16-bit A 16x32 layout)
    const int arow       = lane & 15;
    const int ahalf      = (lane >> 4) & 1;
    const int laneRowOff = ahalf * 8;
    const int ccol       = wv * 16 + (lane & 15);

    const int sr = tid >> 4;             // staging row 0..15
    const int sc = (tid & 15) * 8;       // staging col chunk

#if USE_TDM
    const unsigned long long gtile0 =
        (unsigned long long)(const void*)(x_data + (size_t)row0 * S_ * H_);
    const unsigned int xs0 = (unsigned int)(size_t)(&xstage[0][0]);
    const unsigned int xs1 = (unsigned int)(size_t)(&xstage[1][0]);
    if (wv == 0) tdm_load_tile(gtile0, xs0);             // prefetch t=0
#else
    const float* xbase = x_data + ((size_t)(row0 + sr) * S_) * H_ + sc;
#endif

    // ---------- sequential time loop -----------------------------------------
    for (int t = 0; t < S_; ++t) {
#if USE_TDM
        __builtin_amdgcn_s_wait_tensorcnt(0);  // tile t resident in xstage[t&1]
        __syncthreads();
        {   // convert staged f32 tile -> bf16 xbuf
            const float* xs = &xstage[t & 1][sr * 128 + sc];
            float4 a0 = ((const float4*)xs)[0];
            float4 a1 = ((const float4*)xs)[1];
            __bf16* hp = &xbuf[sr * 128 + sc];
            hp[0] = (__bf16)a0.x; hp[1] = (__bf16)a0.y;
            hp[2] = (__bf16)a0.z; hp[3] = (__bf16)a0.w;
            hp[4] = (__bf16)a1.x; hp[5] = (__bf16)a1.y;
            hp[6] = (__bf16)a1.z; hp[7] = (__bf16)a1.w;
        }
        if (wv == 0 && t + 1 < S_)             // async-load next step's tile
            tdm_load_tile(gtile0 + (size_t)(t + 1) * H_ * sizeof(float),
                          ((t + 1) & 1) ? xs1 : xs0);
        __syncthreads();
#else
        __syncthreads();
        {   // synchronous fallback staging
            const float4* xp = (const float4*)(xbase + (size_t)t * H_);
            float4 a0 = xp[0];
            float4 a1 = xp[1];
            __bf16* hp = &xbuf[sr * 128 + sc];
            hp[0] = (__bf16)a0.x; hp[1] = (__bf16)a0.y;
            hp[2] = (__bf16)a0.z; hp[3] = (__bf16)a0.w;
            hp[4] = (__bf16)a1.x; hp[5] = (__bf16)a1.y;
            hp[6] = (__bf16)a1.z; hp[7] = (__bf16)a1.w;
            if (t + 1 < S_) __builtin_prefetch(xp + (H_ / 4), 0, 0);
        }
        __syncthreads();
#endif

#pragma unroll
        for (int l = 0; l < L_; ++l) {
            const __bf16* Asrc = (l == 0) ? xbuf : hbuf;
            v8f acc1 = {0,0,0,0,0,0,0,0};
            v8f acc2 = {0,0,0,0,0,0,0,0};
            const __bf16* wb1 = &wp[(((l * 16 + wv    ) * 4) * 32 + lane) * 16];
            const __bf16* wb2 = &wp[(((l * 16 + wv + 8) * 4) * 32 + lane) * 16];
#pragma unroll
            for (int kk = 0; kk < 4; ++kk) {
                Frag af, b1, b2;
                const __bf16* ap = &Asrc[arow * 128 + kk * 32 + ahalf * 8];
                af.u[0] = *(const v4u*)(ap);
                af.u[1] = *(const v4u*)(ap + 16);
                const __bf16* p1 = wb1 + kk * (32 * 16);
                const __bf16* p2 = wb2 + kk * (32 * 16);
                b1.u[0] = *(const v4u*)(p1);
                b1.u[1] = *(const v4u*)(p1 + 8);
                b2.u[0] = *(const v4u*)(p2);
                b2.u[1] = *(const v4u*)(p2 + 8);
                acc1 = __builtin_amdgcn_wmma_f32_16x16x32_bf16(
                           false, af.v, false, b1.v, (short)0, acc1, false, false);
                acc2 = __builtin_amdgcn_wmma_f32_16x16x32_bf16(
                           false, af.v, false, b2.v, (short)0, acc2, false, false);
            }
            // gated state update (wave-local)
            float hnew[8];
#pragma unroll
            for (int i = 0; i < 8; ++i) {
                float s1n = s1[l][i] + tanh_fast(acc1[i]);
                float s2o = s2[l][i];
                float s2n = s2o + tanh_fast(acc2[i]);
                float Fu  = fmaxf(s1n, 0.0f);
                hnew[i]   = (1.0f - Fu) * s2n + Fu * s2o;
                s1[l][i] = s1n; s2[l][i] = s2n;
            }
            if (l == 1) __syncthreads();   // WAR: all waves done reading hbuf
#pragma unroll
            for (int i = 0; i < 8; ++i)
                hbuf[(i + laneRowOff) * 128 + ccol] = (__bf16)hnew[i];
            __syncthreads();               // h visible to all waves
        }

        // fc -> softmax[:,1] -> sigmoid -> masked squared loss
        // wave 0 only: lane = row + 16*output; vectorized b128 dot
        if (wv == 0) {
            const int frow = lane & 15;
            const int fo   = lane >> 4;
            float z = 0.0f;
#pragma unroll
            for (int kc = 0; kc < H_; kc += 8) {
                v8bf hv = *(const v8bf*)(&hbuf[frow * 128 + kc]);
                const float4* w4 = (const float4*)(&wfc[fo * H_ + kc]);
                float4 w0 = w4[0], w1 = w4[1];
                z += (float)hv[0]*w0.x + (float)hv[1]*w0.y
                   + (float)hv[2]*w0.z + (float)hv[3]*w0.w
                   + (float)hv[4]*w1.x + (float)hv[5]*w1.y
                   + (float)hv[6]*w1.z + (float)hv[7]*w1.w;
            }
            float zo = __shfl_xor(z, 16, 32);   // pair z0/z1 across half-wave
            if (lane < 16) {                     // z=z0, zo=z1
                float p  = sigmoid_fast(zo - z); // softmax col 1
                float sg = sigmoid_fast(p);
                if (t < len) { float d = label - sg; lossAcc += d * d; }
            }
        }
    }

    if (tid < 16) atomicAdd(out, lossAcc);
}

extern "C" void kernel_launch(void* const* d_in, const int* in_sizes, int n_in,
                              void* d_out, int out_size, void* d_ws, size_t ws_size,
                              hipStream_t stream) {
    const float* x_data   = (const float*)d_in[0];
    const int*   x_length = (const int*)d_in[1];
    const float* x_label  = (const float*)d_in[2];
    const float* Wx2h     = (const float*)d_in[3];
    const float* Wfc      = (const float*)d_in[4];
    float* out = (float*)d_out;

    zero_out_kernel<<<dim3(1), dim3(1), 0, stream>>>(out);
    sru_seq_kernel<<<dim3(B_ / 16), dim3(256), 0, stream>>>(
        x_data, x_length, x_label, Wx2h, Wfc, out);
}
